// CrossAttention_52931176956297
// MI455X (gfx1250) — matrix operand order repaired and checked by reading, not compiled, not run on hardware
//
#include <hip/hip_runtime.h>
#include <hip/hip_bf16.h>

// ---------------------------------------------------------------------------
// Cross-attention for MI455X (gfx1250), wave32 + WMMA f16->f32 + async-LDS.
//   x:(4,4096,512) ctx:(4,1024,768) Wq(512,512) Wk/Wv(768,512) Wo(512,512) bo(512)
// Pipeline:
//   cvt x,ctx -> f16 ; Wq/Wk/Wv/Wo -> f16 transposed (N x K)
//   GEMM (async-LDS B staging): Q = xh@Wq, K = ch@Wk, V = ch@Wv ((b,h,d,key))
//   Flash attention per (b,h,128-q-rows) block, 64-key tiles -> attn f16
//   GEMM: out = attn@Wo + bo (f32 out)
// ---------------------------------------------------------------------------

typedef __attribute__((ext_vector_type(16))) _Float16 v16h;
typedef __attribute__((ext_vector_type(8)))  _Float16 v8h;
typedef __attribute__((ext_vector_type(8)))  float    v8f;
typedef __attribute__((ext_vector_type(4)))  float    v4f;

#define BATCH 4
#define NQ    4096
#define NC    1024
#define DX    512
#define DCTX  768
#define INNER 512
#define NH    8
#define DH    64

union AFrag { v16h v; v8h h[2]; };

// A-fragment 16x32 f16 (row-major source): lane holds row m=lane%16,
// K chunks {base..base+7} and {16+base..16+base+7}, base = (lane>=16)?8:0.
__device__ __forceinline__ v16h load_a16x32(const _Float16* __restrict__ base, int ld) {
  const int lane = threadIdx.x & 31;
  const int m    = lane & 15;
  const int ko   = (lane >> 4) << 3;     // 0 or 8
  const _Float16* p = base + (size_t)m * ld + ko;
  AFrag f;
  f.h[0] = *(const v8h*)(p);             // b128
  f.h[1] = *(const v8h*)(p + 16);        // b128
  return f.v;
}

// B-fragment 32x16 f16 from a transposed (N x K row-major) source: lane holds
// column n=lane%16, K chunks {base..base+7} and {16+base..16+base+7}, base = (lane>=16)?8:0.
__device__ __forceinline__ v16h load_b32x16T(const _Float16* __restrict__ baseT, int ld) {
  const int lane = threadIdx.x & 31;
  const int n    = lane & 15;
  const int ko   = (lane >> 4) << 3;     // 0 or 8
  const _Float16* p = baseT + (size_t)n * ld + ko;
  AFrag f;
  f.h[0] = *(const v8h*)(p);
  f.h[1] = *(const v8h*)(p + 16);
  return f.v;
}

__device__ __forceinline__ v8f wmma_f16(v16h a, v16h b, v8f c) {
  return __builtin_amdgcn_wmma_f32_16x16x32_f16(false, a, false, b, (short)0, c, false, false);
}

// Async copy of 16 bytes global -> LDS (one wave-instruction per 512B tile slice).
__device__ __forceinline__ void async_b128(const _Float16* src, _Float16* lds_dst) {
  const unsigned int       dst = (unsigned int)(uintptr_t)lds_dst;   // addr[31:0] = LDS offset
  const unsigned long long ga  = (unsigned long long)(uintptr_t)src;
  asm volatile("global_load_async_to_lds_b128 %0, %1, off" :: "v"(dst), "v"(ga) : "memory");
}
__device__ __forceinline__ void wait_async0() {
  asm volatile("s_wait_asynccnt 0x0" ::: "memory");
}

// ------------------------------- converters --------------------------------

__global__ __launch_bounds__(256) void cvt_f32_to_f16(const float* __restrict__ in,
                                                      _Float16* __restrict__ out, int n4) {
  int i = blockIdx.x * 256 + threadIdx.x;
  if (i < n4) {
    v4f v = ((const v4f*)in)[i];
    v8h* o = (v8h*)out;                  // write 4 f16 = 8B
    _Float16 h4[4] = {(_Float16)v.x, (_Float16)v.y, (_Float16)v.z, (_Float16)v.w};
    ((ulong1*)o)[i] = *(const ulong1*)h4;
  }
}

// W: K x N (row-major f32)  ->  WT: N x K (row-major f16)
__global__ __launch_bounds__(256) void cvt_transpose_f16(const float* __restrict__ W,
                                                         _Float16* __restrict__ WT,
                                                         int K, int N) {
  int i = blockIdx.x * 256 + threadIdx.x;
  if (i < K * N) {
    int k = i / N, n = i - k * N;
    WT[(size_t)n * K + k] = (_Float16)W[i];
  }
}

// --------------------------------- GEMM ------------------------------------
// C(MxN) = A(MxK,f16) * B(KxN) with B given transposed (NxK,f16).
// Block: 256 thr = 8 waves; block tile 256(M) x 64(N); wave tile 32 x 64.
// B k-slice (64n x 32k, 4KB) double-buffered in LDS via async copies.
// MODE 0: f16 row-major.  MODE 1: f16 V-transposed (b,h,d,key).  MODE 2: f32+bias.

#define BPAD 40   // LDS row length in f16 (80B: 16B aligned, conflict-free b128)

template<int MODE>
__global__ __launch_bounds__(256)
void gemm_wmma_kernel(const _Float16* __restrict__ A, const _Float16* __restrict__ BT,
                      void* __restrict__ Cout, const float* __restrict__ bias,
                      int M, int N, int K) {
  __shared__ __align__(16) _Float16 bsm[2][64][BPAD];

  const int tid  = threadIdx.x;
  const int wave = tid >> 5;
  const int lane = tid & 31;
  const int m0   = blockIdx.x * 256 + wave * 32;
  const int n0   = blockIdx.y * 64;

  // async copy mapping: thread -> 16B chunk of the 64x32 f16 slice
  const int crow = tid >> 2;             // 0..63  (local n)
  const int cchk = tid & 3;              // 0..3   (16B chunk within 32-k row)

  v8f acc0[4] = {}, acc1[4] = {};

  const _Float16* Arow0 = A + (size_t)m0 * K;
  const _Float16* Arow1 = A + (size_t)(m0 + 16) * K;

  // prime buffer 0
  async_b128(BT + (size_t)(n0 + crow) * K + cchk * 8, &bsm[0][crow][cchk * 8]);

  int buf = 0;
  for (int k = 0; k < K; k += 32) {
    wait_async0();
    __syncthreads();
    if (k + 32 < K)
      async_b128(BT + (size_t)(n0 + crow) * K + (k + 32) + cchk * 8,
                 &bsm[buf ^ 1][crow][cchk * 8]);

    __builtin_prefetch(Arow0 + k + 256, 0, 1);
    v16h a0 = load_a16x32(Arow0 + k, K);
    v16h a1 = load_a16x32(Arow1 + k, K);
#pragma unroll
    for (int j = 0; j < 4; ++j) {
      v16h b = load_b32x16T(&bsm[buf][16 * j][0], BPAD);
      acc0[j] = wmma_f16(a0, b, acc0[j]);
      acc1[j] = wmma_f16(a1, b, acc1[j]);
    }
    __syncthreads();
    buf ^= 1;
  }

  const int n  = lane & 15;
  const int mh = (lane >> 4) << 3;       // row offset 0 or 8 in C layout
#pragma unroll
  for (int g = 0; g < 2; ++g) {
#pragma unroll
    for (int j = 0; j < 4; ++j) {
#pragma unroll
      for (int r = 0; r < 8; ++r) {
        const int row = m0 + 16 * g + mh + r;
        const int col = n0 + 16 * j + n;
        const float v = (g == 0) ? acc0[j][r] : acc1[j][r];
        if (MODE == 0) {
          ((_Float16*)Cout)[(size_t)row * N + col] = (_Float16)v;
        } else if (MODE == 1) {
          const int b_  = row >> 10;     // row = b*1024 + key
          const int key = row & 1023;
          const int h_  = col >> 6;      // col = h*64 + d
          const int d_  = col & 63;
          ((_Float16*)Cout)[((((size_t)(b_ * NH + h_)) * DH + d_) << 10) + key] = (_Float16)v;
        } else {
          ((float*)Cout)[(size_t)row * N + col] = v + bias[col];
        }
      }
    }
  }
}

// ----------------------------- flash attention -----------------------------
// grid.x = NQ/128, grid.y = B*H. Block 256 thr = 8 waves, wave = 16 q-rows.
// 64-key tiles: 8 WMMA scores + 8 WMMA PV per tile per wave.
// Qh,Kh: (b, seq, inner) f16.  Vt: (b, h, d, key) f16.  Oh: (b, q, inner) f16.

#define PPAD 72   // LDS P row length in f16 (144B: 16B aligned, conflict-free b128)

__global__ __launch_bounds__(256)
void flash_attn_kernel(const _Float16* __restrict__ Qh, const _Float16* __restrict__ Kh,
                       const _Float16* __restrict__ Vt, _Float16* __restrict__ Oh) {
  __shared__ __align__(16) _Float16 pbuf[8][16][PPAD];   // ~18KB

  const int wave = threadIdx.x >> 5;
  const int lane = threadIdx.x & 31;
  const int hi   = lane >> 4;            // lane half
  const int nl   = lane & 15;
  const int b    = blockIdx.y >> 3;
  const int h    = blockIdx.y & 7;
  const int q0   = blockIdx.x * 128 + wave * 16;

  const _Float16* Qbase = Qh + ((size_t)(b * NQ + q0)) * INNER + h * DH;
  const _Float16* Kb    = Kh + ((size_t)b * NC) * INNER + h * DH;
  const _Float16* Vb    = Vt + ((size_t)(b * NH + h)) * DH * NC;

  // Q A-fragments (d 0..31 and 32..63), kept in VGPRs for the whole kernel.
  const v16h qa0 = load_a16x32(Qbase, INNER);
  const v16h qa1 = load_a16x32(Qbase + 32, INNER);

  v8f o0 = {}, o1 = {}, o2 = {}, o3 = {};
  float mrow[8], lrow[8];
#pragma unroll
  for (int r = 0; r < 8; ++r) { mrow[r] = -1e30f; lrow[r] = 0.0f; }

  // softmax(s * Dh^-0.5) via exp2: scale scores by 0.125*log2(e)
  const float sscale = 0.125f * 1.44269504088896340736f;

  for (int kt = 0; kt < NC; kt += 64) {
    // ---- scores: four 16x16 subtiles over 64 keys, K-dim = 64 (2 WMMA each)
    v8f s0 = {}, s1 = {}, s2 = {}, s3 = {};
    s0 = wmma_f16(qa0, load_b32x16T(Kb + (size_t)(kt +  0) * INNER, INNER), s0);
    s0 = wmma_f16(qa1, load_b32x16T(Kb + (size_t)(kt +  0) * INNER + 32, INNER), s0);
    s1 = wmma_f16(qa0, load_b32x16T(Kb + (size_t)(kt + 16) * INNER, INNER), s1);
    s1 = wmma_f16(qa1, load_b32x16T(Kb + (size_t)(kt + 16) * INNER + 32, INNER), s1);
    s2 = wmma_f16(qa0, load_b32x16T(Kb + (size_t)(kt + 32) * INNER, INNER), s2);
    s2 = wmma_f16(qa1, load_b32x16T(Kb + (size_t)(kt + 32) * INNER + 32, INNER), s2);
    s3 = wmma_f16(qa0, load_b32x16T(Kb + (size_t)(kt + 48) * INNER, INNER), s3);
    s3 = wmma_f16(qa1, load_b32x16T(Kb + (size_t)(kt + 48) * INNER + 32, INNER), s3);

    // ---- online softmax (per C-fragment row; reduce over 16-lane halves)
    float alpha[8];
#pragma unroll
    for (int r = 0; r < 8; ++r) {
      const float a0 = s0[r] * sscale;
      const float a1 = s1[r] * sscale;
      const float a2 = s2[r] * sscale;
      const float a3 = s3[r] * sscale;
      float tm = fmaxf(fmaxf(a0, a1), fmaxf(a2, a3));
      tm = fmaxf(tm, __shfl_xor(tm, 1, 32));
      tm = fmaxf(tm, __shfl_xor(tm, 2, 32));
      tm = fmaxf(tm, __shfl_xor(tm, 4, 32));
      tm = fmaxf(tm, __shfl_xor(tm, 8, 32));
      const float mnew = fmaxf(mrow[r], tm);
      alpha[r] = __builtin_exp2f(mrow[r] - mnew);
      mrow[r]  = mnew;
      const float p0 = __builtin_exp2f(a0 - mnew);
      const float p1 = __builtin_exp2f(a1 - mnew);
      const float p2 = __builtin_exp2f(a2 - mnew);
      const float p3 = __builtin_exp2f(a3 - mnew);
      float ts = (p0 + p1) + (p2 + p3);
      ts += __shfl_xor(ts, 1, 32);
      ts += __shfl_xor(ts, 2, 32);
      ts += __shfl_xor(ts, 4, 32);
      ts += __shfl_xor(ts, 8, 32);
      lrow[r] = lrow[r] * alpha[r] + ts;
      // stage P (C layout) into LDS for re-layout into A fragments
      const int row = r + 8 * hi;
      pbuf[wave][row][nl]      = (_Float16)p0;
      pbuf[wave][row][16 + nl] = (_Float16)p1;
      pbuf[wave][row][32 + nl] = (_Float16)p2;
      pbuf[wave][row][48 + nl] = (_Float16)p3;
    }

    // ---- rescale running O by alpha (per row)
#pragma unroll
    for (int r = 0; r < 8; ++r) {
      o0[r] *= alpha[r]; o1[r] *= alpha[r]; o2[r] *= alpha[r]; o3[r] *= alpha[r];
    }

    // ---- P as two 16x32 A-fragments (wave-private LDS, DS in-order per wave)
    const v16h pa0 = load_a16x32(&pbuf[wave][0][0], PPAD);    // keys kt..kt+31
    const v16h pa1 = load_a16x32(&pbuf[wave][0][32], PPAD);   // keys kt+32..kt+63

    // ---- O += P * V_tile ; Vt rows = d, contiguous keys
    o0 = wmma_f16(pa0, load_b32x16T(Vb + (size_t)( 0) * NC + kt, NC), o0);
    o0 = wmma_f16(pa1, load_b32x16T(Vb + (size_t)( 0) * NC + kt + 32, NC), o0);
    o1 = wmma_f16(pa0, load_b32x16T(Vb + (size_t)(16) * NC + kt, NC), o1);
    o1 = wmma_f16(pa1, load_b32x16T(Vb + (size_t)(16) * NC + kt + 32, NC), o1);
    o2 = wmma_f16(pa0, load_b32x16T(Vb + (size_t)(32) * NC + kt, NC), o2);
    o2 = wmma_f16(pa1, load_b32x16T(Vb + (size_t)(32) * NC + kt + 32, NC), o2);
    o3 = wmma_f16(pa0, load_b32x16T(Vb + (size_t)(48) * NC + kt, NC), o3);
    o3 = wmma_f16(pa1, load_b32x16T(Vb + (size_t)(48) * NC + kt + 32, NC), o3);
  }

  // ---- finalize: divide by row sums, store f16 (b, q, h*64+d)
  _Float16* Ob = Oh + ((size_t)(b * NQ + q0)) * INNER + h * DH;
#pragma unroll
  for (int r = 0; r < 8; ++r) {
    const float inv = 1.0f / lrow[r];
    const int row = r + 8 * hi;
    Ob[(size_t)row * INNER + 0 * 16 + nl] = (_Float16)(o0[r] * inv);
    Ob[(size_t)row * INNER + 1 * 16 + nl] = (_Float16)(o1[r] * inv);
    Ob[(size_t)row * INNER + 2 * 16 + nl] = (_Float16)(o2[r] * inv);
    Ob[(size_t)row * INNER + 3 * 16 + nl] = (_Float16)(o3[r] * inv);
  }
}

// ------------------------------- launcher ----------------------------------

extern "C" void kernel_launch(void* const* d_in, const int* in_sizes, int n_in,
                              void* d_out, int out_size, void* d_ws, size_t ws_size,
                              hipStream_t stream) {
  const float* x   = (const float*)d_in[0];
  const float* ctx = (const float*)d_in[1];
  const float* Wq  = (const float*)d_in[2];
  const float* Wk  = (const float*)d_in[3];
  const float* Wv  = (const float*)d_in[4];
  const float* Wo  = (const float*)d_in[5];
  const float* bo  = (const float*)d_in[6];
  float* out = (float*)d_out;

  char* ws = (char*)d_ws;
  size_t off = 0;
  auto alloc_h = [&](size_t elems) { _Float16* p = (_Float16*)(ws + off); off += elems * 2; return p; };

  _Float16* xh   = alloc_h((size_t)BATCH * NQ * DX);       // 16.8 MB
  _Float16* ch   = alloc_h((size_t)BATCH * NC * DCTX);     //  6.3 MB
  _Float16* WqT  = alloc_h((size_t)DX * INNER);            //  0.5 MB (N x K)
  _Float16* WkT  = alloc_h((size_t)DCTX * INNER);          //  0.8 MB
  _Float16* WvT  = alloc_h((size_t)DCTX * INNER);          //  0.8 MB
  _Float16* WoT  = alloc_h((size_t)INNER * INNER);         //  0.5 MB
  _Float16* Qh   = alloc_h((size_t)BATCH * NQ * INNER);    // 16.8 MB
  _Float16* Kh   = alloc_h((size_t)BATCH * NC * INNER);    //  4.2 MB
  _Float16* Vt   = alloc_h((size_t)BATCH * NH * DH * NC);  //  4.2 MB (b,h,d,key)
  _Float16* attn = alloc_h((size_t)BATCH * NQ * INNER);    // 16.8 MB

  // 1) down-convert activations (vectorized x4)
  {
    int n4 = (BATCH * NQ * DX) / 4;
    cvt_f32_to_f16<<<(n4 + 255) / 256, 256, 0, stream>>>(x, xh, n4);
    n4 = (BATCH * NC * DCTX) / 4;
    cvt_f32_to_f16<<<(n4 + 255) / 256, 256, 0, stream>>>(ctx, ch, n4);
  }
  // 2) transpose + down-convert weights (to N x K)
  cvt_transpose_f16<<<(DX * INNER + 255) / 256, 256, 0, stream>>>(Wq, WqT, DX, INNER);
  cvt_transpose_f16<<<(DCTX * INNER + 255) / 256, 256, 0, stream>>>(Wk, WkT, DCTX, INNER);
  cvt_transpose_f16<<<(DCTX * INNER + 255) / 256, 256, 0, stream>>>(Wv, WvT, DCTX, INNER);
  cvt_transpose_f16<<<(INNER * INNER + 255) / 256, 256, 0, stream>>>(Wo, WoT, INNER, INNER);

  // 3) projections (block tile 256x64)
  {
    dim3 gq((BATCH * NQ) / 256, INNER / 64);
    gemm_wmma_kernel<0><<<gq, 256, 0, stream>>>(xh, WqT, Qh, nullptr, BATCH * NQ, INNER, DX);
    dim3 gk((BATCH * NC) / 256, INNER / 64);
    gemm_wmma_kernel<0><<<gk, 256, 0, stream>>>(ch, WkT, Kh, nullptr, BATCH * NC, INNER, DCTX);
    gemm_wmma_kernel<1><<<gk, 256, 0, stream>>>(ch, WvT, Vt, nullptr, BATCH * NC, INNER, DCTX);
  }

  // 4) flash attention
  {
    dim3 ga(NQ / 128, BATCH * NH);
    flash_attn_kernel<<<ga, 256, 0, stream>>>(Qh, Kh, Vt, attn);
  }

  // 5) output projection + bias -> f32 d_out
  {
    dim3 go((BATCH * NQ) / 256, INNER / 64);
    gemm_wmma_kernel<2><<<go, 256, 0, stream>>>(attn, WoT, out, bo, BATCH * NQ, INNER, INNER);
  }
}
